// HyperspectralCmKANLayer_26912265076905
// MI455X (gfx1250) — matrix operand order repaired and checked by reading, not compiled, not run on hardware
//
#include <hip/hip_runtime.h>

typedef __attribute__((ext_vector_type(16))) _Float16 v16h;
typedef __attribute__((ext_vector_type(8)))  _Float16 v8h;
typedef __attribute__((ext_vector_type(8)))  float    v8f;

#define IN_C 31
#define OUT_C 31
#define HW 16384        // 128*128 pixels
#define UW_OFF 7688
#define RW_OFF 8649
// fragment storage: per (i, ob, f) one 16x32 f16 A-fragment = 32 lanes * 16 halfs
#define FRAGS_PER_I 20          // 2 ob * (8 coef + uw + rw)
#define NFRAG (31 * FRAGS_PER_I)
#define FRAG_HALFS 512
#define FRAG_BYTES 1024
#define STAGE_BYTES (FRAGS_PER_I * FRAG_BYTES)   // 20480 per i
// dynamic LDS layout: [stage0 20480][stage1 20480][basis 64*31*16][silu 64*31*4]
#define SMEM_BASIS_OFF (2 * STAGE_BYTES)
#define SMEM_SILU_OFF  (SMEM_BASIS_OFF + 64 * IN_C * 16)
#define SMEM_TOTAL     (SMEM_SILU_OFF + 64 * IN_C * 4)   // 80,640 B (<320KB WGP LDS)

// A-matrix 16-bit 16x32 element->K mapping (CDNA5 ISA 7.12.2):
// VGPR v holds K pair; v<4 -> K = 2v + pos + 8*half, v>=4 -> K = 16 + 2(v-4) + pos + 8*half
__device__ __forceinline__ int c_of_A(int j, int hi) {
    int v = j >> 1, pos = j & 1;
    int base = (v < 4) ? (v * 2) : (16 + (v - 4) * 2);
    return base + pos + hi * 8;
}
// B-matrix 16-bit 32x16 mapping: same K order as A (j<8 -> K = j + 8*half, j>=8 -> K = 16 + (j-8) + 8*half).
__device__ __forceinline__ int c_of_B(int j, int hi) {
    return c_of_A(j, hi);
}

// ---------------- kernel 1: repack gen_w/gen_b into f16 WMMA A-fragments ----------------
__global__ __launch_bounds__(256) void kan_pack(const float* __restrict__ gen_w,
                                                const float* __restrict__ gen_b,
                                                _Float16* __restrict__ wsA) {
    int tid = blockIdx.x * 256 + threadIdx.x;
    if (tid >= NFRAG * FRAG_HALFS) return;
    int frag = tid >> 9;
    int e    = tid & 511;
    int lane = e >> 4;
    int j    = e & 15;
    int i    = frag / FRAGS_PER_I;
    int rem  = frag % FRAGS_PER_I;
    int ob   = rem / 10;
    int f    = rem % 10;
    int m  = lane & 15;
    int hi = lane >> 4;
    int c  = c_of_A(j, hi);
    int o  = ob * 16 + m;
    float val = 0.0f;
    if (o < OUT_C) {
        int p;
        if (f < 8)       p = i * 248 + o * 8 + f;     // coef: ((i*31+o)*8+k)
        else if (f == 8) p = UW_OFF + i * 31 + o;     // univariate weight
        else             p = RW_OFF + i * 31 + o;     // residual weight
        val = (c < IN_C) ? gen_w[p * IN_C + c] : gen_b[p];  // bias folded as channel 31
    }
    wsA[(size_t)frag * FRAG_HALFS + lane * 16 + j] = (_Float16)val;
}

// ---------------- B-spline basis (grid=5, order=3, range [-1,1]) ----------------
__device__ __forceinline__ void bspline8(float xv, float* b8) {
    const float h = 0.4f;
    float t[12];
#pragma unroll
    for (int j = 0; j < 12; ++j) t[j] = (float)(j - 3) * h - 1.0f;
    float b[11];
#pragma unroll
    for (int j = 0; j < 11; ++j) b[j] = (xv >= t[j] && xv < t[j + 1]) ? 1.0f : 0.0f;
#pragma unroll
    for (int p = 1; p <= 3; ++p) {
        float rinv = 1.0f / ((float)p * h);   // uniform knots: all denominators = p*h
#pragma unroll
        for (int j = 0; j < 10; ++j) {
            if (j < 11 - p) {
                float left  = (xv - t[j]) * rinv * b[j];
                float right = (t[j + p + 1] - xv) * rinv * b[j + 1];
                b[j] = left + right;
            }
        }
    }
#pragma unroll
    for (int k = 0; k < 8; ++k) b8[k] = b[k];
}

// ---------------- kernel 2: fused generator-GEMM + KAN via WMMA ----------------
// 4 waves / block, each wave owns 16 pixels; 256 blocks cover 16384 pixels.
// Double-buffered async weight staging (L2 -> LDS) overlapped with WMMA compute.
__global__ __launch_bounds__(128) void kan_main(const float* __restrict__ x,
                                                const _Float16* __restrict__ wsA,
                                                float* __restrict__ out) {
    extern __shared__ __align__(32) char smem[];
    v8h*   basis_sh = (v8h*)(smem + SMEM_BASIS_OFF);
    float* silu_sh  = (float*)(smem + SMEM_SILU_OFF);

    const int lane = threadIdx.x & 31;
    const int wave = threadIdx.x >> 5;
    const int n    = lane & 15;
    const int hi   = lane >> 4;
    const int pix  = blockIdx.x * 64 + wave * 16 + n;
    const int ploc = wave * 16 + n;
    const char* srcb = (const char*)wsA;

    // prefetch weight stage for i=0 while we do the prepass (buffer 0 = smem base)
    {
        unsigned lds_base = (unsigned)(unsigned long long)(char*)smem;
#pragma unroll
        for (int r = 0; r < 10; ++r) {
            int off = (r * 128 + (int)threadIdx.x) * 16;
            unsigned lds = lds_base + (unsigned)off;
            unsigned long long g = (unsigned long long)(srcb + off);
            asm volatile("global_load_async_to_lds_b128 %0, %1, off"
                         :: "v"(lds), "v"(g) : "memory");
        }
    }

    // xfe B-fragment: channels as K (31 real + constant 1.0 bias channel)
    v16h xfe;
#pragma unroll
    for (int j = 0; j < 16; ++j) {
        int c = c_of_B(j, hi);
        xfe[j] = (c < IN_C) ? (_Float16)x[c * HW + pix] : (_Float16)1.0f;
    }

    // cooperative basis + silu precompute: lane-half parity splits channels
    for (int it = 0; it < 16; ++it) {
        int i = it * 2 + hi;
        if (i < IN_C) {
            float xv = x[i * HW + pix];
            float b8[8];
            bspline8(xv, b8);
            v8h bh;
#pragma unroll
            for (int k = 0; k < 8; ++k) bh[k] = (_Float16)b8[k];
            basis_sh[ploc * IN_C + i] = bh;
            silu_sh[ploc * IN_C + i]  = xv / (1.0f + __expf(-xv));
        }
    }
    __syncthreads();

    v8f y0 = {}, y1 = {};

    for (int i = 0; i < IN_C; ++i) {
        // ping-pong buffers computed arithmetically (no LDS-pointer aggregates)
        const _Float16* cur = (const _Float16*)(smem + (i & 1) * STAGE_BYTES);
        // issue next iteration's weight stage into the other buffer
        if (i + 1 < IN_C) {
            char* nxt = smem + ((i + 1) & 1) * STAGE_BYTES;
            unsigned lds_base = (unsigned)(unsigned long long)nxt;
#pragma unroll
            for (int r = 0; r < 10; ++r) {
                int off = (r * 128 + (int)threadIdx.x) * 16;
                unsigned lds = lds_base + (unsigned)off;
                unsigned long long g =
                    (unsigned long long)(srcb + (size_t)(i + 1) * STAGE_BYTES + off);
                asm volatile("global_load_async_to_lds_b128 %0, %1, off"
                             :: "v"(lds), "v"(g) : "memory");
            }
            // async loads complete in order: <=10 outstanding means cur's 10 landed
            asm volatile("s_wait_asynccnt 0xa" ::: "memory");
        } else {
            asm volatile("s_wait_asynccnt 0x0" ::: "memory");
        }
        __syncthreads();   // all waves' copies of cur landed

        // per-lane basis / silu for (pixel n, channel i)
        v8h bh = basis_sh[ploc * IN_C + i];
        float sil = silu_sh[ploc * IN_C + i];
        // B-fragments for the spline GEMM: xfe scaled by basis_k (uniform per fragment)
        v16h Bu[8];
#pragma unroll
        for (int s = 0; s < 8; ++s) Bu[s] = xfe * bh[s];

#pragma unroll
        for (int ob = 0; ob < 2; ++ob) {
            const _Float16* base = cur + (ob * 10) * FRAG_HALFS;
            // split accumulator chains: 4+ independent WMMA streams hide the
            // F16-WMMA RAW hazard window (5 slots) without v_nops
            v8f accA = {}, accB = {};
#pragma unroll
            for (int s = 0; s < 8; s += 2) {
                v16h Aa = *(const v16h*)(base + s * FRAG_HALFS + lane * 16);
                v16h Ab = *(const v16h*)(base + (s + 1) * FRAG_HALFS + lane * 16);
                accA = __builtin_amdgcn_wmma_f32_16x16x32_f16(
                    false, Aa, false, Bu[s], (short)0, accA, false, false);
                accB = __builtin_amdgcn_wmma_f32_16x16x32_f16(
                    false, Ab, false, Bu[s + 1], (short)0, accB, false, false);
            }
            v16h Auw = *(const v16h*)(base + 8 * FRAG_HALFS + lane * 16);
            v16h Arw = *(const v16h*)(base + 9 * FRAG_HALFS + lane * 16);
            v8f zc = {};
            v8f uwd = __builtin_amdgcn_wmma_f32_16x16x32_f16(
                false, Auw, false, xfe, (short)0, zc, false, false);
            v8f rwd = __builtin_amdgcn_wmma_f32_16x16x32_f16(
                false, Arw, false, xfe, (short)0, zc, false, false);
            // identical D layouts -> pure register elementwise epilogue
            if (ob == 0) {
#pragma unroll
                for (int r = 0; r < 8; ++r)
                    y0[r] += (accA[r] + accB[r]) * uwd[r] + sil * rwd[r];
            } else {
#pragma unroll
                for (int r = 0; r < 8; ++r)
                    y1[r] += (accA[r] + accB[r]) * uwd[r] + sil * rwd[r];
            }
        }
        __syncthreads();   // all waves done reading cur before it is overwritten (i+2)
    }

    // D layout: component r, lane-half hi -> output channel r + 8*hi (+16 for ob=1)
#pragma unroll
    for (int r = 0; r < 8; ++r) {
        int o = r + hi * 8;
        out[o * HW + pix] = y0[r];
        int o2 = 16 + r + hi * 8;
        if (o2 < OUT_C) out[o2 * HW + pix] = y1[r];
    }
}

extern "C" void kernel_launch(void* const* d_in, const int* in_sizes, int n_in,
                              void* d_out, int out_size, void* d_ws, size_t ws_size,
                              hipStream_t stream) {
    const float* x     = (const float*)d_in[0];
    const float* gen_w = (const float*)d_in[1];
    const float* gen_b = (const float*)d_in[2];
    float* out         = (float*)d_out;
    _Float16* wsA      = (_Float16*)d_ws;   // needs NFRAG*FRAG_BYTES = 634,880 bytes

    int pack_threads = NFRAG * FRAG_HALFS;                 // 317,440
    kan_pack<<<(pack_threads + 255) / 256, 256, 0, stream>>>(gen_w, gen_b, wsA);
    kan_main<<<HW / 64, 128, SMEM_TOTAL, stream>>>(x, wsA, out);
    (void)in_sizes; (void)n_in; (void)out_size; (void)ws_size;
}
